// FlowTransformerBlock_32504312496489
// MI455X (gfx1250) — matrix operand order repaired and checked by reading, not compiled, not run on hardware
//
#include <hip/hip_runtime.h>
#include <hip/hip_bf16.h>
#include <cstdint>

typedef __attribute__((ext_vector_type(16))) _Float16 v16h;
typedef __attribute__((ext_vector_type(8)))  _Float16 v8h;
typedef __attribute__((ext_vector_type(8)))  float    v8f;

#define HH 256
#define WW 256
#define HWN (HH*WW)
#define CC 128
#define K9 1152   // 128 * 9

// ---------------------------------------------------------------------------
// CDNA5 async global->LDS copy (ASYNCcnt-tracked), 16 bytes per lane.
// Low 32 bits of a generic shared pointer are the LDS byte address.
// ---------------------------------------------------------------------------
__device__ __forceinline__ void async_copy_b128(void* ldsp, const void* gp) {
  unsigned l = (unsigned)(uintptr_t)ldsp;
  asm volatile("global_load_async_to_lds_b128 %0, %1, off"
               :: "v"(l), "v"(gp) : "memory");
}
__device__ __forceinline__ void wait_async() {
  asm volatile("s_wait_asynccnt 0x0" ::: "memory");
}

// ---------------------------------------------------------------------------
// Repack OIHW f32 weights -> f16 [cout][tap*128 + cin] for contiguous A-frags
// ---------------------------------------------------------------------------
__global__ __launch_bounds__(256)
void pack_w_kernel(const float* __restrict__ W, _Float16* __restrict__ Wt) {
  int i = blockIdx.x * 256 + threadIdx.x;       // over 128*128*9
  if (i >= CC * CC * 9) return;
  int kx = i % 3; int t1 = i / 3;
  int ky = t1 % 3; int t2 = t1 / 3;
  int ci = t2 % CC; int o = t2 / CC;
  Wt[(size_t)o * K9 + (ky * 3 + kx) * CC + ci] = (_Float16)W[i];
}

// ---------------------------------------------------------------------------
// Layout convert: f32 NCHW -> f16 NHWC via 64px x 128ch LDS transpose tile
// (both global sides coalesced).
// ---------------------------------------------------------------------------
__global__ __launch_bounds__(256)
void cvt_nhwc_kernel(const float* __restrict__ in, _Float16* __restrict__ outh) {
  __shared__ _Float16 T[64 * 128];
  const int tid = threadIdx.x;
  const int px0 = blockIdx.x * 64;
  const int img = blockIdx.y;
  const size_t ibase = (size_t)img * CC * HWN;
  for (int e = tid; e < 64 * 128; e += 256) {      // read: px-inner (coalesced)
    int ch = e >> 6, p = e & 63;
    T[p * 128 + ch] = (_Float16)in[ibase + (size_t)ch * HWN + px0 + p];
  }
  __syncthreads();
  const size_t obase = ((size_t)img * HWN + px0) * CC;
  for (int e = tid; e < 64 * 128; e += 256) {      // write: ch-inner (coalesced)
    int p = e >> 7, ch = e & 127;
    outh[obase + (size_t)p * CC + ch] = T[p * 128 + ch];
  }
}

// ---------------------------------------------------------------------------
// Fused: out = inF + prelu(conv3x3(inH) + bias, alpha)
// inH: f16 NHWC (async-staged), inF: f32 NCHW (residual).
// Implicit GEMM, 8 waves = 8 Cout tiles x 16 pixels, WMMA f32_16x16x32_f16.
// ---------------------------------------------------------------------------
__global__ __launch_bounds__(256)
void conv3_prelu_res_kernel(const _Float16* __restrict__ inH,
                            const float* __restrict__ inF,
                            const _Float16* __restrict__ Wt,
                            const float* __restrict__ bias,
                            const float* __restrict__ alpha,
                            float* __restrict__ outf,
                            _Float16* __restrict__ outh) {
  __shared__ _Float16 sm[3 * 18 * 128];            // halo [pos][cin], 13.5 KB
  const int tid = threadIdx.x;
  const int x0  = blockIdx.x * 16;
  const int y   = blockIdx.y;
  const int img = blockIdx.z;

  // Async-stage halo: 54 positions x 256B, 16B segments; zero-fill OOB.
  for (int i = tid; i < 54 * 16; i += 256) {
    int pos = i >> 4, seg = i & 15;
    int r = pos / 18, cc = pos - r * 18;
    int yy = y + r - 1, xx = x0 + cc - 1;
    _Float16* lp = &sm[pos * 128 + seg * 8];
    if (yy >= 0 && yy < HH && xx >= 0 && xx < WW) {
      const _Float16* gp = inH + ((size_t)img * HWN + (size_t)yy * WW + xx) * CC + seg * 8;
      async_copy_b128(lp, gp);
    } else {
      *(int4*)lp = int4{0, 0, 0, 0};
    }
  }
  wait_async();
  __syncthreads();

  const int lane  = tid & 31;
  const int wave  = tid >> 5;
  const int hsel  = lane >> 4;
  const int l15   = lane & 15;
  const int coutB = wave * 16;

  v8f acc = {};
  for (int step = 0; step < 36; ++step) {
    const int tap = step >> 2;            // ky*3+kx
    const int c0  = (step & 3) * 32;
    const _Float16* wp = Wt + (size_t)(coutB + l15) * K9 + tap * 128 + c0 + hsel * 16;
    __builtin_prefetch(wp + 128, 0, 1);
    v16h a;
    v8h alo = *(const v8h*)wp;            // K base..base+7
    v8h ahi = *(const v8h*)(wp + 8);      // K base+8..base+15
    for (int i = 0; i < 8; ++i) { a[i] = alo[i]; a[i + 8] = ahi[i]; }
    const int ky = tap / 3, kx = tap - ky * 3;
    const _Float16* bp = &sm[(ky * 18 + (l15 + kx)) * 128 + c0 + hsel * 16];
    v16h bf = *(const v16h*)bp;
    acc = __builtin_amdgcn_wmma_f32_16x16x32_f16(false, a, false, bf,
                                                 (short)0, acc, false, false);
  }

  const float al = alpha[0];
  const int px = x0 + l15;
  const float* ibF = inF + (size_t)img * CC * HWN;
  for (int j = 0; j < 8; ++j) {
    int co = coutB + hsel * 8 + j;
    float v = acc[j] + bias[co];
    v = (v >= 0.0f) ? v : al * v;
    float res = ibF[(size_t)co * HWN + (size_t)y * WW + px] + v;
    if (outf) outf[(size_t)(img * CC + co) * HWN + (size_t)y * WW + px] = res;
    if (outh) outh[((size_t)img * HWN + (size_t)y * WW + px) * CC + co] = (_Float16)res;
  }
}

// ---------------------------------------------------------------------------
// Dual-branch windowed attention, one block per 8x8 window.
// Q/K rows are f16 NHWC -> async-staged. Key chunk buffer aliased:
// [key][ch] during S, transposed [ch][key] during O (contiguous B-frags).
// ---------------------------------------------------------------------------
__global__ __launch_bounds__(256)
void win_attn_kernel(const _Float16* __restrict__ QhN, const float* __restrict__ Qf,
                     const _Float16* __restrict__ KVhN, const _Float16* __restrict__ KVwhN,
                     const float* __restrict__ em,
                     float* __restrict__ Xo, _Float16* __restrict__ XoH) {
  __shared__ _Float16 Qs[64 * 128];   // 16 KB  [q][c]
  __shared__ _Float16 KbU[32 * 128];  //  8 KB  S: [key][c] / O: [c][key]
  __shared__ _Float16 Sb[64 * 256];   // 32 KB  [q][key]

  const int tid = threadIdx.x;
  const int blk = blockIdx.x;
  const int b   = blk >> 10;
  const int wr  = blk & 1023;
  const int oy  = (wr >> 5) * 8, ox = (wr & 31) * 8;

  // Async-stage Q window: 64 rows x 256B
  for (int i = tid; i < 64 * 16; i += 256) {
    int q = i >> 4, seg = i & 15;
    int y = oy + (q >> 3), x = ox + (q & 7);
    async_copy_b128(&Qs[q * 128 + seg * 8],
                    QhN + ((size_t)b * HWN + (size_t)y * WW + x) * CC + seg * 8);
  }

  const int lane = tid & 31, wave = tid >> 5;
  const int hsel = lane >> 4, l15 = lane & 15;
  const float scale = 0.088388347648318447f;   // 128^-0.5

  v8f o1[4] = {}, o2[4] = {};

  for (int br = 0; br < 2; ++br) {
    const int nkv = br ? 256 : 128;
    const _Float16* kv = br ? KVwhN : KVhN;
    const int imgBase = b * (br ? 4 : 2);

    // ---- S = Q K^T ----
    for (int k0 = 0; k0 < nkv; k0 += 32) {
      wait_async();
      __syncthreads();
      for (int i = tid; i < 32 * 16; i += 256) {   // 32 keys x 256B, async
        int klocal = i >> 4, seg = i & 15;
        int kk = k0 + klocal;
        int tt = kk >> 6, qq = kk & 63;
        int y = oy + (qq >> 3), x = ox + (qq & 7);
        async_copy_b128(&KbU[klocal * 128 + seg * 8],
                        kv + ((size_t)(imgBase + tt) * HWN + (size_t)y * WW + x) * CC + seg * 8);
      }
      wait_async();
      __syncthreads();
      const int mt = wave >> 1, ntl = wave & 1;
      v8f sacc = {};
      for (int ks = 0; ks < 4; ++ks) {
        const int c0 = ks * 32;
        const _Float16* ap = &Qs[(mt * 16 + l15) * 128 + c0 + hsel * 16];
        v16h a; v8h alo = *(const v8h*)ap, ahi = *(const v8h*)(ap + 8);
        for (int i = 0; i < 8; ++i) { a[i] = alo[i]; a[i + 8] = ahi[i]; }
        const _Float16* bp = &KbU[(ntl * 16 + l15) * 128 + c0 + hsel * 16];
        v16h bf = *(const v16h*)bp;
        sacc = __builtin_amdgcn_wmma_f32_16x16x32_f16(false, a, false, bf,
                                                      (short)0, sacc, false, false);
      }
      for (int j = 0; j < 8; ++j) {
        int q = mt * 16 + hsel * 8 + j;
        int key = k0 + ntl * 16 + l15;
        Sb[q * 256 + key] = (_Float16)(sacc[j] * scale);
      }
    }
    __syncthreads();

    // ---- softmax, one row per thread ----
    if (tid < 64) {
      _Float16* rowp = &Sb[tid * 256];
      float mx = -1e30f;
      for (int k = 0; k < nkv; ++k) { float v = (float)rowp[k]; mx = v > mx ? v : mx; }
      float sum = 0.0f;
      for (int k = 0; k < nkv; ++k) {
        float e = __expf((float)rowp[k] - mx);
        sum += e; rowp[k] = (_Float16)e;
      }
      float inv = 1.0f / sum;
      for (int k = 0; k < nkv; ++k) rowp[k] = (_Float16)((float)rowp[k] * inv);
    }

    // ---- O = P K (key chunk restaged TRANSPOSED: KbT[ch][key]) ----
    v8f oac[4] = {};
    for (int k0 = 0; k0 < nkv; k0 += 32) {
      __syncthreads();   // softmax done / KbU reuse
      for (int i = tid; i < 2048; i += 256) {      // b32 ch-pairs, coalesced reads
        int chp = i & 63, klocal = i >> 6;
        int kk = k0 + klocal;
        int tt = kk >> 6, qq = kk & 63;
        int y = oy + (qq >> 3), x = ox + (qq & 7);
        const _Float16* gp =
            kv + ((size_t)(imgBase + tt) * HWN + (size_t)y * WW + x) * CC + chp * 2;
        _Float16 h0 = gp[0], h1 = gp[1];
        KbU[(chp * 2) * 32 + klocal] = h0;
        KbU[(chp * 2 + 1) * 32 + klocal] = h1;
      }
      __syncthreads();
      for (int i = 0; i < 4; ++i) {               // 32 tiles / 8 waves
        const int id = wave * 4 + i;
        const int mt = id >> 3, nt = id & 7;
        const _Float16* ap = &Sb[(mt * 16 + l15) * 256 + k0 + hsel * 16];
        v16h a; v8h alo = *(const v8h*)ap, ahi = *(const v8h*)(ap + 8);
        for (int ii = 0; ii < 8; ++ii) { a[ii] = alo[ii]; a[ii + 8] = ahi[ii]; }
        const int ch = nt * 16 + l15;
        v16h bf = *(const v16h*)&KbU[ch * 32 + hsel * 16];  // contiguous K 0..15/16..31
        oac[i] = __builtin_amdgcn_wmma_f32_16x16x32_f16(false, a, false, bf,
                                                        (short)0, oac[i], false, false);
      }
    }
    if (br == 0) { for (int i = 0; i < 4; ++i) o1[i] = oac[i]; }
    else         { for (int i = 0; i < 4; ++i) o2[i] = oac[i]; }
  }

  // ---- mask blend + shortcut; write f32 NCHW and f16 NHWC ----
  for (int i = 0; i < 4; ++i) {
    const int id = wave * 4 + i;
    const int mt = id >> 3, nt = id & 7;
    const int ch = nt * 16 + l15;
    for (int j = 0; j < 8; ++j) {
      int q = mt * 16 + hsel * 8 + j;
      int y = oy + (q >> 3), x = ox + (q & 7);
      float m = em[(size_t)b * HWN + (size_t)y * WW + x];
      size_t idx = (size_t)(b * CC + ch) * HWN + (size_t)y * WW + x;
      float r = Qf[idx] + m * o1[i][j] + (1.0f - m) * o2[i][j];
      Xo[idx] = r;
      XoH[((size_t)b * HWN + (size_t)y * WW + x) * CC + ch] = (_Float16)r;
    }
  }
}

// ---------------------------------------------------------------------------
extern "C" void kernel_launch(void* const* d_in, const int* in_sizes, int n_in,
                              void* d_out, int out_size, void* d_ws, size_t ws_size,
                              hipStream_t stream) {
  (void)in_sizes; (void)n_in; (void)out_size; (void)ws_size;
  const float* xq   = (const float*)d_in[0];
  const float* xkvw = (const float*)d_in[1];
  const float* xkv  = (const float*)d_in[2];
  const float* em   = (const float*)d_in[3];
  const float* Wq   = (const float*)d_in[4];
  const float* bq   = (const float*)d_in[5];
  const float* aq   = (const float*)d_in[6];
  const float* Wkv  = (const float*)d_in[7];
  const float* bkv  = (const float*)d_in[8];
  const float* akv  = (const float*)d_in[9];
  const float* Wff  = (const float*)d_in[10];
  const float* bff  = (const float*)d_in[11];
  const float* aff  = (const float*)d_in[12];
  float* out = (float*)d_out;

  char* ws = (char*)d_ws;
  size_t off = 0;
  auto carve = [&](size_t bytes) -> void* {
    void* p = ws + off;
    off = (off + bytes + 255) & ~(size_t)255;
    return p;
  };
  const size_t imgF32 = (size_t)CC * HWN * 4;
  const size_t imgF16 = (size_t)CC * HWN * 2;
  _Float16* wq16  = (_Float16*)carve((size_t)CC * K9 * 2);
  _Float16* wkv16 = (_Float16*)carve((size_t)CC * K9 * 2);
  _Float16* wff16 = (_Float16*)carve((size_t)CC * K9 * 2);
  _Float16* XqH   = (_Float16*)carve(2 * imgF16);   // inputs, f16 NHWC
  _Float16* XkvH  = (_Float16*)carve(4 * imgF16);
  _Float16* XkvwH = (_Float16*)carve(8 * imgF16);
  float*    Qf    = (float*)   carve(2 * imgF32);   // Q shortcut, f32 NCHW
  _Float16* Qh    = (_Float16*)carve(2 * imgF16);   // Q f16 NHWC
  _Float16* KVh   = (_Float16*)carve(4 * imgF16);   // KV f16 NHWC
  _Float16* KVwh  = (_Float16*)carve(8 * imgF16);   // KVw f16 NHWC
  float*    Xo    = (float*)   carve(2 * imgF32);   // attn out + shortcut, f32 NCHW
  _Float16* XoH   = (_Float16*)carve(2 * imgF16);   // same, f16 NHWC

  // weight repack
  pack_w_kernel<<<576, 256, 0, stream>>>(Wq,  wq16);
  pack_w_kernel<<<576, 256, 0, stream>>>(Wkv, wkv16);
  pack_w_kernel<<<576, 256, 0, stream>>>(Wff, wff16);

  // activations -> f16 NHWC (enables async global->LDS staging)
  cvt_nhwc_kernel<<<dim3(HWN / 64, 2), 256, 0, stream>>>(xq,   XqH);
  cvt_nhwc_kernel<<<dim3(HWN / 64, 4), 256, 0, stream>>>(xkv,  XkvH);
  cvt_nhwc_kernel<<<dim3(HWN / 64, 8), 256, 0, stream>>>(xkvw, XkvwH);

  // residual conv embeddings
  conv3_prelu_res_kernel<<<dim3(WW / 16, HH, 2), 256, 0, stream>>>(
      XqH, xq, wq16, bq, aq, Qf, Qh);
  conv3_prelu_res_kernel<<<dim3(WW / 16, HH, 4), 256, 0, stream>>>(
      XkvH, xkv, wkv16, bkv, akv, nullptr, KVh);
  conv3_prelu_res_kernel<<<dim3(WW / 16, HH, 8), 256, 0, stream>>>(
      XkvwH, xkvw, wkv16, bkv, akv, nullptr, KVwh);

  // mask-blended dual-branch windowed attention + shortcut
  win_attn_kernel<<<2048, 256, 0, stream>>>(Qh, Qf, KVh, KVwh, em, Xo, XoH);

  // feed-forward residual conv -> final output
  conv3_prelu_res_kernel<<<dim3(WW / 16, HH, 2), 256, 0, stream>>>(
      XoH, Xo, wff16, bff, aff, out, nullptr);
}
